// ASTEncoderLayer_20710332301705
// MI455X (gfx1250) — matrix operand order repaired and checked by reading, not compiled, not run on hardware
//
#include <hip/hip_runtime.h>
#include <hip/hip_bf16.h>

// ---------------------------------------------------------------- types
typedef __bf16 bf16_t;
typedef bf16_t v8bf  __attribute__((ext_vector_type(8)));
typedef bf16_t v16bf __attribute__((ext_vector_type(16)));
typedef float  v8f   __attribute__((ext_vector_type(8)));

#define DMODEL 512
#define NHEADS 8
#define DHEAD  64
#define DFF    2048
#define NBUCK  64
#define BB     4
#define LL     1024
#define MTOT   (BB * LL)   // 4096 rows

// ---------------------------------------------------------------- helpers
__device__ __forceinline__ bf16_t f2bf(float f) {
  unsigned u = __builtin_bit_cast(unsigned, f);
  unsigned r = u + 0x7FFFu + ((u >> 16) & 1u);            // RNE
  unsigned short h = (unsigned short)(r >> 16);
  return __builtin_bit_cast(bf16_t, h);
}

__device__ __forceinline__ v16bf cat8(v8bf lo, v8bf hi) {
  return __builtin_shufflevector(lo, hi, 0,1,2,3,4,5,6,7,8,9,10,11,12,13,14,15);
}

// A-operand fragment (16x32 bf16): rows striped over lanes, K packed per lane.
__device__ __forceinline__ v16bf frag_a(const bf16_t* base, int r0, int ld, int koff) {
  int lane = threadIdx.x & 31;
  int kb = (lane < 16) ? 0 : 8;
  const bf16_t* p = base + (r0 + (lane & 15)) * ld + koff;
  v8bf lo = *(const v8bf*)(p + kb);
  v8bf hi = *(const v8bf*)(p + 16 + kb);
  return cat8(lo, hi);
}

// B-operand fragment (32x16 bf16) from a tile stored [n][k] (row = output col n).
__device__ __forceinline__ v16bf frag_b(const bf16_t* base, int n0, int ld, int koff) {
  int lane = threadIdx.x & 31;
  const bf16_t* p = base + (n0 + (lane & 15)) * ld + koff + ((lane < 16) ? 0 : 8);
  v8bf lo = *(const v8bf*)(p);
  v8bf hi = *(const v8bf*)(p + 16);
  return cat8(lo, hi);
}

__device__ __forceinline__ v8f wmma_bf16(v16bf a, v16bf b, v8f c) {
  return __builtin_amdgcn_wmma_f32_16x16x32_bf16(false, a, false, b, (short)0, c,
                                                 false, false);
}

// Single-instruction DPP reductions within 16-lane rows (wave32):
// v_max/v_add with src0 routed through the DPP network: d = op(perm(x), x).
// xor1/xor2/xor7/xor8 generate the full 16-lane group; the two 16-lane halves
// (lanes 0-15 vs 16-31) stay independent, matching the C-matrix layout.
__device__ __forceinline__ float red_max16(float x) {
  float t;
  asm("v_max_num_f32_dpp %0, %1, %1 quad_perm:[1,0,3,2] row_mask:0xf bank_mask:0xf bound_ctrl:1"
      : "=v"(t) : "v"(x));
  x = t;
  asm("v_max_num_f32_dpp %0, %1, %1 quad_perm:[2,3,0,1] row_mask:0xf bank_mask:0xf bound_ctrl:1"
      : "=v"(t) : "v"(x));
  x = t;
  asm("v_max_num_f32_dpp %0, %1, %1 row_half_mirror row_mask:0xf bank_mask:0xf bound_ctrl:1"
      : "=v"(t) : "v"(x));
  x = t;
  asm("v_max_num_f32_dpp %0, %1, %1 row_ror:8 row_mask:0xf bank_mask:0xf bound_ctrl:1"
      : "=v"(t) : "v"(x));
  return t;
}
__device__ __forceinline__ float red_sum16(float x) {
  float t;
  asm("v_add_f32_dpp %0, %1, %1 quad_perm:[1,0,3,2] row_mask:0xf bank_mask:0xf bound_ctrl:1"
      : "=v"(t) : "v"(x));
  x = t;
  asm("v_add_f32_dpp %0, %1, %1 quad_perm:[2,3,0,1] row_mask:0xf bank_mask:0xf bound_ctrl:1"
      : "=v"(t) : "v"(x));
  x = t;
  asm("v_add_f32_dpp %0, %1, %1 row_half_mirror row_mask:0xf bank_mask:0xf bound_ctrl:1"
      : "=v"(t) : "v"(x));
  x = t;
  asm("v_add_f32_dpp %0, %1, %1 row_ror:8 row_mask:0xf bank_mask:0xf bound_ctrl:1"
      : "=v"(t) : "v"(x));
  return t;
}

__device__ __forceinline__ float gelu_tanh(float x) {
  float x3 = x * x * x;
  return 0.5f * x * (1.0f + tanhf(0.7978845608028654f * (x + 0.044715f * x3)));
}

// ---------------------------------------------------------------- layernorm -> bf16
__global__ void ln_bf16_kernel(const float* __restrict__ x, const float* __restrict__ g,
                               const float* __restrict__ b, bf16_t* __restrict__ out) {
  const int D = DMODEL;
  int row = blockIdx.x;
  int t = threadIdx.x;            // 128 threads
  const float* xr = x + (size_t)row * D;
  float vals[4], s = 0.f, s2 = 0.f;
#pragma unroll
  for (int i = 0; i < 4; i++) {
    float v = xr[t + i * 128];
    vals[i] = v; s += v; s2 += v * v;
  }
  __shared__ float red0[128], red1[128];
  red0[t] = s; red1[t] = s2;
  __syncthreads();
  for (int off = 64; off > 0; off >>= 1) {
    if (t < off) { red0[t] += red0[t + off]; red1[t] += red1[t + off]; }
    __syncthreads();
  }
  float mu = red0[0] * (1.0f / D);
  float var = red1[0] * (1.0f / D) - mu * mu;
  float rinv = rsqrtf(var + 1e-5f);
#pragma unroll
  for (int i = 0; i < 4; i++) {
    int c = t + i * 128;
    out[(size_t)row * D + c] = f2bf((vals[i] - mu) * rinv * g[c] + b[c]);
  }
}

// ---------------------------------------------------------------- generic WMMA GEMM
// C[M,N] = act(A[M,K](bf16) * B(f32) + bias) + resid ; 64x64 tile, 4 waves,
// K-step 32, register-staged double buffering (next tile's global loads overlap
// with current tile's WMMAs).
// TB=0: B stored [K,N] (weights); TB=1: B stored [N,K] (rel tables -> implicit B^T).
template <bool TB, bool BIAS, bool ACT, bool RES, bool OBF>
__global__ void gemm_wmma_kernel(const bf16_t* __restrict__ A, const float* __restrict__ B,
                                 const float* __restrict__ bias,
                                 const float* __restrict__ resid, void* __restrict__ Cout,
                                 int K, int lda, int ldb, int ldc, int zdiv,
                                 long sA0, long sA1, long sB0, long sB1, long sC0, long sC1) {
  __shared__ __align__(16) bf16_t As[64 * 32];
  __shared__ __align__(16) bf16_t Bs[64 * 32];   // [n][k]

  int z = blockIdx.z;
  int zb = z / zdiv, zh = z - zb * zdiv;
  A += zb * sA0 + zh * sA1;
  B += zb * sB0 + zh * sB1;
  long coff = zb * sC0 + zh * sC1;

  int tid = threadIdx.x;
  int w = tid >> 5, lane = tid & 31;
  int wy = w >> 1, wx = w & 1;
  int m0 = blockIdx.y * 64, n0 = blockIdx.x * 64;

  // per-thread staging coordinates
  int arow = tid >> 2, acol = (tid & 3) << 3;              // A: 2 chunks (row, row+32)
  v8bf aR[2];
  float4 bR[4];

  v8f acc[2][2];
#pragma unroll
  for (int i = 0; i < 2; i++)
#pragma unroll
    for (int j = 0; j < 2; j++)
#pragma unroll
      for (int e = 0; e < 8; e++) acc[i][j][e] = 0.f;

  // ---- global tile load into registers (no LDS touch)
  auto load_tile = [&](int kt) {
    aR[0] = *(const v8bf*)(A + (size_t)(m0 + arow) * lda + kt + acol);
    aR[1] = *(const v8bf*)(A + (size_t)(m0 + arow + 32) * lda + kt + acol);
#pragma unroll
    for (int i = 0; i < 4; i++) {
      int c = tid + i * 128;
      if (TB) {
        int n = c >> 3, kk = (c & 7) << 2;
        bR[i] = *(const float4*)(B + (size_t)(n0 + n) * ldb + kt + kk);
      } else {
        int kk = c >> 4, n4 = (c & 15) << 2;
        bR[i] = *(const float4*)(B + (size_t)(kt + kk) * ldb + n0 + n4);
      }
    }
  };
  // ---- registers -> LDS
  auto store_tile = [&]() {
    *(v8bf*)&As[arow * 32 + acol] = aR[0];
    *(v8bf*)&As[(arow + 32) * 32 + acol] = aR[1];
#pragma unroll
    for (int i = 0; i < 4; i++) {
      int c = tid + i * 128;
      if (TB) {
        int n = c >> 3, kk = (c & 7) << 2;
        Bs[n * 32 + kk + 0] = f2bf(bR[i].x);
        Bs[n * 32 + kk + 1] = f2bf(bR[i].y);
        Bs[n * 32 + kk + 2] = f2bf(bR[i].z);
        Bs[n * 32 + kk + 3] = f2bf(bR[i].w);
      } else {
        int kk = c >> 4, n4 = (c & 15) << 2;
        Bs[(n4 + 0) * 32 + kk] = f2bf(bR[i].x);
        Bs[(n4 + 1) * 32 + kk] = f2bf(bR[i].y);
        Bs[(n4 + 2) * 32 + kk] = f2bf(bR[i].z);
        Bs[(n4 + 3) * 32 + kk] = f2bf(bR[i].w);
      }
    }
  };

  load_tile(0);
  for (int kt = 0; kt < K; kt += 32) {
    __syncthreads();             // previous iteration's fragment reads done
    store_tile();
    __syncthreads();
    if (kt + 32 < K) load_tile(kt + 32);   // overlap with WMMAs below

    v16bf a0 = frag_a(As, wy * 32, 32, 0);
    v16bf a1 = frag_a(As, wy * 32 + 16, 32, 0);
    v16bf b0 = frag_b(Bs, wx * 32, 32, 0);
    v16bf b1 = frag_b(Bs, wx * 32 + 16, 32, 0);
    acc[0][0] = wmma_bf16(a0, b0, acc[0][0]);
    acc[0][1] = wmma_bf16(a0, b1, acc[0][1]);
    acc[1][0] = wmma_bf16(a1, b0, acc[1][0]);
    acc[1][1] = wmma_bf16(a1, b1, acc[1][1]);
  }

  // epilogue: C layout => lanes 0-15: N=lane, M=v ; lanes 16-31: N=lane-16, M=v+8
  int mrow = (lane < 16) ? 0 : 8;
  int nc = lane & 15;
#pragma unroll
  for (int fy = 0; fy < 2; fy++)
#pragma unroll
    for (int fx = 0; fx < 2; fx++)
#pragma unroll
      for (int v = 0; v < 8; v++) {
        int gm = m0 + wy * 32 + fy * 16 + mrow + v;
        int gn = n0 + wx * 32 + fx * 16 + nc;
        float val = acc[fy][fx][v];
        if (BIAS) val += bias[gn];
        if (ACT)  val = gelu_tanh(val);
        if (RES)  val += resid[(size_t)gm * ldc + gn];
        size_t idx = coff + (size_t)gm * ldc + gn;
        if (OBF) ((bf16_t*)Cout)[idx] = f2bf(val);
        else     ((float*)Cout)[idx] = val;
      }
}

// ---------------------------------------------------------------- flash attention
// grid: (L/64, B*H); block 128 (4 waves, 16 i-rows each). Online softmax over j.
// launch_bounds(128,1): occupancy is grid-limited, let the allocator use full
// VGPR budget (kills the few-dword loop spill seen at default settings).
__global__ void __launch_bounds__(128, 1)
attn_wmma_kernel(const bf16_t* __restrict__ q, const bf16_t* __restrict__ k,
                 const bf16_t* __restrict__ v,
                 const float* __restrict__ qrk, const float* __restrict__ krq,
                 const int* __restrict__ pidx,
                 const unsigned char* __restrict__ mask,
                 bf16_t* __restrict__ out) {
  const int L = LL, D = DMODEL, Dh = DHEAD;
  int bh = blockIdx.y;
  int b = bh >> 3, h = bh & 7;
  int w = threadIdx.x >> 5, lane = threadIdx.x & 31;
  int i0 = blockIdx.x * 64 + w * 16;

  __shared__ __align__(16) bf16_t Qs[4][16 * 64];  // per-wave Q tile [i][d]
  __shared__ __align__(16) bf16_t Ks[32 * 64];     // [j][d]
  __shared__ __align__(16) bf16_t Vst[64 * 32];    // [d][j]  (transposed)
  __shared__ __align__(16) bf16_t Ps[4][16 * 32];  // per-wave P staging

  const size_t bL = (size_t)b * L;
  const bf16_t* qb = q + bL * D + h * Dh;
  const bf16_t* kb = k + bL * D + h * Dh;
  const bf16_t* vb = v + bL * D + h * Dh;

  int mrow = (lane < 16) ? 0 : 8;
  int nc = lane & 15;

  // per-lane gather bases (constant parts fold into instruction offsets)
  const int* pRow = pidx + (bL + i0 + mrow) * L + nc;             // [vv*L + j]
  const float* qrow = qrk + ((size_t)bh * L + i0 + mrow) * NBUCK; // [vv*64 + bucket]
  const float* krqb = krq + (size_t)bh * L * NBUCK;               // [j*64 + bucket]

  // stage this wave's Q tile into LDS (private region; same-wave DS is in-order)
#pragma unroll
  for (int c = lane; c < 128; c += 32) {
    int row = c >> 3, dd = (c & 7) << 3;
    *(v8bf*)&Qs[w][row * 64 + dd] = *(const v8bf*)(qb + (size_t)(i0 + row) * D + dd);
  }

  v8f o[4];
  float rm[8], rs[8];
#pragma unroll
  for (int d = 0; d < 4; d++)
#pragma unroll
    for (int e = 0; e < 8; e++) o[d][e] = 0.f;
#pragma unroll
  for (int e = 0; e < 8; e++) { rm[e] = -3.0e38f; rs[e] = 0.f; }

  for (int j0 = 0; j0 < L; j0 += 32) {
    // stage K (row-major) and V (transposed) tiles
#pragma unroll
    for (int c = threadIdx.x; c < 256; c += 128) {
      int jj = c >> 3, dd = (c & 7) << 3;
      *(v8bf*)&Ks[jj * 64 + dd] = *(const v8bf*)(kb + (size_t)(j0 + jj) * D + dd);
    }
#pragma unroll
    for (int c = threadIdx.x; c < 256; c += 128) {
      int jj = c >> 3, dd = (c & 7) << 3;
      v8bf vv8 = *(const v8bf*)(vb + (size_t)(j0 + jj) * D + dd);
#pragma unroll
      for (int e = 0; e < 8; e++) Vst[(dd + e) * 32 + jj] = vv8[e];
    }

    // batch the bias gathers for maximum memory-level parallelism:
    // 16 independent idx loads, then 32 independent table loads.
    int ka[8], kb2[8];
#pragma unroll
    for (int vv = 0; vv < 8; vv++) {
      ka[vv]  = pRow[vv * L + j0];
      kb2[vv] = pRow[vv * L + j0 + 16];
    }
    int jA = j0 + nc, jB = j0 + 16 + nc;
    bool mA = mask[bL + jA] != 0;
    bool mB = mask[bL + jB] != 0;
    const float* krA = krqb + (size_t)jA * NBUCK;
    const float* krB = krqb + (size_t)jB * NBUCK;
    float gA[8], gB[8];
#pragma unroll
    for (int vv = 0; vv < 8; vv++) {
      gA[vv] = qrow[vv * NBUCK + ka[vv]]  + krA[ka[vv]];
      gB[vv] = qrow[vv * NBUCK + kb2[vv]] + krB[kb2[vv]];
    }

    __syncthreads();

    // S tiles: c0 = cols j0..j0+15, c1 = cols j0+16..j0+31 (K over d = 64)
    v8f c0, c1;
#pragma unroll
    for (int e = 0; e < 8; e++) { c0[e] = 0.f; c1[e] = 0.f; }
    {
      v16bf qa0 = frag_a(Qs[w], 0, 64, 0);
      v16bf qa1 = frag_a(Qs[w], 0, 64, 32);
      c0 = wmma_bf16(qa0, frag_b(Ks, 0, 64, 0), c0);
      c0 = wmma_bf16(qa1, frag_b(Ks, 0, 64, 32), c0);
      c1 = wmma_bf16(qa0, frag_b(Ks, 16, 64, 0), c1);
      c1 = wmma_bf16(qa1, frag_b(Ks, 16, 64, 32), c1);
    }

    // online softmax, one row (vv) at a time (gathers already in registers)
#pragma unroll
    for (int vv = 0; vv < 8; vv++) {
      float sA = (c0[vv] + gA[vv]) * 0.125f;
      float sB = (c1[vv] + gB[vv]) * 0.125f;
      sA = mA ? sA : -1e9f;
      sB = mB ? sB : -1e9f;
      float lm = red_max16(fmaxf(sA, sB));
      float nm = fmaxf(rm[vv], lm);
      float fac = __expf(rm[vv] - nm);
      rm[vv] = nm;
      float pA = __expf(sA - nm);
      float pB = __expf(sB - nm);
      float ls = red_sum16(pA + pB);
      rs[vv] = rs[vv] * fac + ls;
      o[0][vv] *= fac; o[1][vv] *= fac; o[2][vv] *= fac; o[3][vv] *= fac;
      Ps[w][(mrow + vv) * 32 + nc] = f2bf(pA);
      Ps[w][(mrow + vv) * 32 + 16 + nc] = f2bf(pB);
    }

    // O += P(16x32) x V(32x64)  (P re-read in A layout; same-wave LDS is in-order)
    v16bf pa = frag_a(Ps[w], 0, 32, 0);
    o[0] = wmma_bf16(pa, frag_b(Vst, 0, 32, 0), o[0]);
    o[1] = wmma_bf16(pa, frag_b(Vst, 16, 32, 0), o[1]);
    o[2] = wmma_bf16(pa, frag_b(Vst, 32, 32, 0), o[2]);
    o[3] = wmma_bf16(pa, frag_b(Vst, 48, 32, 0), o[3]);
    __syncthreads();
  }

  // normalize and store (heads concatenated: col = h*64 + d)
  bf16_t* ob = out + bL * D + h * Dh;
#pragma unroll
  for (int db = 0; db < 4; db++)
#pragma unroll
    for (int vv = 0; vv < 8; vv++) {
      float val = o[db][vv] / rs[vv];
      ob[(size_t)(i0 + mrow + vv) * D + db * 16 + nc] = f2bf(val);
    }
}

// ---------------------------------------------------------------- launch
extern "C" void kernel_launch(void* const* d_in, const int* in_sizes, int n_in,
                              void* d_out, int out_size, void* d_ws, size_t ws_size,
                              hipStream_t stream) {
  (void)in_sizes; (void)n_in; (void)out_size; (void)ws_size;
  const float* src  = (const float*)d_in[0];
  const int* pidx   = (const int*)d_in[1];
  const unsigned char* mask = (const unsigned char*)d_in[2];
  const float* rel_q = (const float*)d_in[3];
  const float* rel_k = (const float*)d_in[4];
  const float* Wq = (const float*)d_in[5];  const float* bq = (const float*)d_in[6];
  const float* Wk = (const float*)d_in[7];  const float* bk = (const float*)d_in[8];
  const float* Wv = (const float*)d_in[9];  const float* bv = (const float*)d_in[10];
  const float* Wo = (const float*)d_in[11]; const float* bo = (const float*)d_in[12];
  const float* W1 = (const float*)d_in[13]; const float* b1 = (const float*)d_in[14];
  const float* W2 = (const float*)d_in[15]; const float* b2 = (const float*)d_in[16];
  const float* ln1g = (const float*)d_in[17]; const float* ln1b = (const float*)d_in[18];
  const float* ln2g = (const float*)d_in[19]; const float* ln2b = (const float*)d_in[20];
  float* out = (float*)d_out;

  char* w = (char*)d_ws;
  bf16_t* h1   = (bf16_t*)w; w += (size_t)MTOT * DMODEL * 2;            // 4 MB
  bf16_t* qb   = (bf16_t*)w; w += (size_t)MTOT * DMODEL * 2;            // 4 MB
  bf16_t* kb   = (bf16_t*)w; w += (size_t)MTOT * DMODEL * 2;            // 4 MB
  bf16_t* vb   = (bf16_t*)w; w += (size_t)MTOT * DMODEL * 2;            // 4 MB
  float*  qrk  = (float*)w;  w += (size_t)BB * NHEADS * LL * NBUCK * 4; // 8 MB
  float*  krq  = (float*)w;  w += (size_t)BB * NHEADS * LL * NBUCK * 4; // 8 MB
  bf16_t* attb = (bf16_t*)w; w += (size_t)MTOT * DMODEL * 2;            // 4 MB
  float*  x    = (float*)w;  w += (size_t)MTOT * DMODEL * 4;            // 8 MB
  bf16_t* h2   = (bf16_t*)w; w += (size_t)MTOT * DMODEL * 2;            // 4 MB
  bf16_t* ff1  = (bf16_t*)w; w += (size_t)MTOT * DFF * 2;               // 16 MB

  dim3 blk(128);

  // 1) ln1
  ln_bf16_kernel<<<MTOT, blk, 0, stream>>>(src, ln1g, ln1b, h1);

  // 2) QKV projections (bf16 out)
  dim3 gq(DMODEL / 64, MTOT / 64, 1);
  gemm_wmma_kernel<false, true, false, false, true><<<gq, blk, 0, stream>>>(
      h1, Wq, bq, nullptr, qb, DMODEL, DMODEL, DMODEL, DMODEL, 1, 0, 0, 0, 0, 0, 0);
  gemm_wmma_kernel<false, true, false, false, true><<<gq, blk, 0, stream>>>(
      h1, Wk, bk, nullptr, kb, DMODEL, DMODEL, DMODEL, DMODEL, 1, 0, 0, 0, 0, 0, 0);
  gemm_wmma_kernel<false, true, false, false, true><<<gq, blk, 0, stream>>>(
      h1, Wv, bv, nullptr, vb, DMODEL, DMODEL, DMODEL, DMODEL, 1, 0, 0, 0, 0, 0, 0);

  // 3) rel-position projections: qrk = q @ rel_k^T, krq = k @ rel_q^T  (batched B*H)
  dim3 gr(NBUCK / 64, LL / 64, BB * NHEADS);
  gemm_wmma_kernel<true, false, false, false, false><<<gr, blk, 0, stream>>>(
      qb, rel_k, nullptr, nullptr, qrk, DHEAD, DMODEL, DHEAD, NBUCK, NHEADS,
      (long)LL * DMODEL, DHEAD, 0, 0, (long)NHEADS * LL * NBUCK, (long)LL * NBUCK);
  gemm_wmma_kernel<true, false, false, false, false><<<gr, blk, 0, stream>>>(
      kb, rel_q, nullptr, nullptr, krq, DHEAD, DMODEL, DHEAD, NBUCK, NHEADS,
      (long)LL * DMODEL, DHEAD, 0, 0, (long)NHEADS * LL * NBUCK, (long)LL * NBUCK);

  // 4) flash attention with relative bias
  dim3 ga(LL / 64, BB * NHEADS);
  attn_wmma_kernel<<<ga, blk, 0, stream>>>(qb, kb, vb, qrk, krq, pidx, mask, attb);

  // 5) output projection + residual(src) -> x (f32)
  gemm_wmma_kernel<false, true, false, true, false><<<gq, blk, 0, stream>>>(
      attb, Wo, bo, src, x, DMODEL, DMODEL, DMODEL, DMODEL, 1, 0, 0, 0, 0, 0, 0);

  // 6) ln2
  ln_bf16_kernel<<<MTOT, blk, 0, stream>>>(x, ln2g, ln2b, h2);

  // 7) FF1 + GELU (bf16 out)
  dim3 g1(DFF / 64, MTOT / 64, 1);
  gemm_wmma_kernel<false, true, true, false, true><<<g1, blk, 0, stream>>>(
      h2, W1, b1, nullptr, ff1, DMODEL, DMODEL, DFF, DFF, 1, 0, 0, 0, 0, 0, 0);

  // 8) FF2 + residual(x) -> out (f32)
  gemm_wmma_kernel<false, true, false, true, false><<<gq, blk, 0, stream>>>(
      ff1, W2, b2, x, out, DFF, DFF, DMODEL, DMODEL, 1, 0, 0, 0, 0, 0, 0);
}
